// GRU_with_GCC_29987461660953
// MI455X (gfx1250) — matrix operand order repaired and checked by reading, not compiled, not run on hardware
//
#include <hip/hip_runtime.h>
#include <math.h>

// ---------------------------------------------------------------------------
// GRU with gated context features — MI455X (gfx1250, wave32, WMMA bf16)
//   B=64, S=256, D=H=F0=F1=1024
//   Phase A: gi = X·Wih^T + b_ih            [16384 x 3072]   (WMMA bf16)
//            gf = F0·Wf0^T + F1·Wf1^T + b   [16384 x 2048]   (WMMA bf16)
//   Phase B: persistent recurrence kernel, 256 steps, global spin barrier,
//            double-buffered hidden state, WMMA bf16 for h·Whh^T, with the
//            block's W_hh slice LDS-resident (96 KB of the WGP's 320 KB).
// ---------------------------------------------------------------------------

typedef __attribute__((ext_vector_type(16))) __bf16 v16bf;
typedef __attribute__((ext_vector_type(8)))  float  v8f;

#define H_DIM 1024
#define B_DIM 64
#define S_DIM 256
#define BS    (B_DIM * S_DIM)   // 16384 rows for the batched projections

__device__ __forceinline__ unsigned short f2bf(float f) {
  unsigned u = __float_as_uint(f);
  unsigned r = 0x7FFFu + ((u >> 16) & 1u);      // round-to-nearest-even
  return (unsigned short)((u + r) >> 16);
}

union FragU { uint4 q[2]; v16bf v; };

// A-matrix 16x32 bf16 fragment from row-major [*, ld] bf16 storage.
// Per ISA: lane m = lane&15; element e -> K = 16*(e>=8) + 8*(lane>=16) + (e&7)
// => two contiguous 16-byte runs per lane.
__device__ __forceinline__ v16bf load_a_frag(const unsigned short* __restrict__ A,
                                             int ld, int row0, int k0, int lane) {
  const int m  = row0 + (lane & 15);
  const int kb = (lane >> 4) << 3;              // 0 or 8
  const unsigned short* p = A + (size_t)m * ld + k0 + kb;
  FragU u;
  u.q[0] = *reinterpret_cast<const uint4*>(p);
  u.q[1] = *reinterpret_cast<const uint4*>(p + 16);
  return u.v;
}

// B-matrix 32x16 bf16 fragment from pre-packed fragment-linear storage:
// one contiguous 32-byte load per lane (1 KB contiguous per wave).
__device__ __forceinline__ v16bf load_b_frag(const unsigned short* __restrict__ P,
                                             long frag, int lane) {
  const unsigned short* p = P + frag * 512 + lane * 16;
  FragU u;
  u.q[0] = reinterpret_cast<const uint4*>(p)[0];
  u.q[1] = reinterpret_cast<const uint4*>(p)[1];
  return u.v;
}

__device__ __forceinline__ v8f wmma_bf16(v16bf a, v16bf b, v8f c) {
  // (neg_a, A, neg_b, B, c_mod, C, reuse_a, reuse_b)
  return __builtin_amdgcn_wmma_f32_16x16x32_bf16(false, a, false, b,
                                                 (short)0, c, false, false);
}

// ---------------------------------------------------------------------------
// Prep kernels
// ---------------------------------------------------------------------------

__global__ void init_state(float* __restrict__ hF, unsigned short* __restrict__ hB,
                           unsigned* __restrict__ ctr) {
  const int i = blockIdx.x * blockDim.x + threadIdx.x;   // 131072 threads
  if (i < 2 * B_DIM * H_DIM) { hF[i] = 0.0f; hB[i] = 0; }
  if (i == 0) *ctr = 0u;
}

__global__ void f32_to_bf16_k(const float* __restrict__ x,
                              unsigned short* __restrict__ y, long n) {
  long i = (long)blockIdx.x * blockDim.x + threadIdx.x;
  const long stride = (long)gridDim.x * blockDim.x;
  for (; i < n; i += stride) y[i] = f2bf(x[i]);
}

// Pack row-major fp32 weight W[N,K] (used as B = W^T, so B[k][n] = W[n][k])
// into WMMA B-fragment layout: frag = nTile*kTiles + kTile; per lane:
// n = nTile*16 + (lane&15); K = kTile*32 + 16*(e>=8) + 8*(lane>=16) + (e&7), e=0..15.
__global__ void pack_weights(const float* __restrict__ W,
                             unsigned short* __restrict__ out, int N, int K) {
  const int  kT    = K >> 5;
  const long total = (long)(N >> 4) * kT * 32;
  long idx = (long)blockIdx.x * blockDim.x + threadIdx.x;
  if (idx >= total) return;
  const int  lane = (int)(idx & 31);
  const long frag = idx >> 5;
  const int  kt   = (int)(frag % kT);
  const int  nt   = (int)(frag / kT);
  const int  n    = (nt << 4) + (lane & 15);
  const int  k0   = (kt << 5) + ((lane >> 4) << 3);
  const float* src = W + (size_t)n * K + k0;
  unsigned short* dst = out + frag * 512 + lane * 16;
#pragma unroll
  for (int e = 0; e < 16; ++e) dst[e] = f2bf(src[e < 8 ? e : e + 8]);
}

// ---------------------------------------------------------------------------
// Phase A: big projection GEMM  OUT[M,Nout] = A1·W1^T (+ A2·W2^T) + bias
// Block = 256 thr (8 waves), tile 128x128; wave tile 32x64 (2x4 frags).
// ---------------------------------------------------------------------------

__global__ __launch_bounds__(256) void gemm_bf16_nt(
    const unsigned short* __restrict__ A1, const unsigned short* __restrict__ W1,
    const unsigned short* __restrict__ A2, const unsigned short* __restrict__ W2,
    const float* __restrict__ bias1, const float* __restrict__ bias2,
    float* __restrict__ OUT, int M, int Nout, int K)
{
  const int lane = threadIdx.x & 31;
  const int wave = threadIdx.x >> 5;
  const int wr   = wave >> 1;                 // 0..3
  const int wc   = wave & 1;                  // 0..1
  const int row0 = blockIdx.y * 128 + wr * 32;
  const int col0 = blockIdx.x * 128 + wc * 64;
  const int kT   = K >> 5;

  v8f acc[2][4];
#pragma unroll
  for (int i = 0; i < 2; ++i)
#pragma unroll
    for (int j = 0; j < 4; ++j) acc[i][j] = (v8f){0,0,0,0,0,0,0,0};

  for (int src = 0; src < 2; ++src) {
    const unsigned short* A = src ? A2 : A1;
    const unsigned short* W = src ? W2 : W1;
    if (A == nullptr) continue;               // uniform branch
    for (int kt = 0; kt < kT; ++kt) {
      const int k0 = kt << 5;
      v16bf a0 = load_a_frag(A, K, row0,      k0, lane);
      v16bf a1 = load_a_frag(A, K, row0 + 16, k0, lane);
#pragma unroll
      for (int cf = 0; cf < 4; ++cf) {
        v16bf b = load_b_frag(W, (long)((col0 >> 4) + cf) * kT + kt, lane);
        acc[0][cf] = wmma_bf16(a0, b, acc[0][cf]);
        acc[1][cf] = wmma_bf16(a1, b, acc[1][cf]);
      }
    }
  }

  // C/D layout: elem v of lane l -> M = 8*(l>=16)+v, N = l&15
  const int nb = col0 + (lane & 15);
  const int rb = row0 + ((lane >> 4) << 3);
#pragma unroll
  for (int cf = 0; cf < 4; ++cf) {
    const int n = nb + cf * 16;
    float bsum = 0.0f;
    if (bias1) bsum += bias1[n];
    if (bias2) bsum += bias2[n];
#pragma unroll
    for (int rf = 0; rf < 2; ++rf) {
#pragma unroll
      for (int v = 0; v < 8; ++v) {
        OUT[(size_t)(rb + rf * 16 + v) * Nout + n] = acc[rf][cf][v] + bsum;
      }
    }
  }
}

// ---------------------------------------------------------------------------
// Phase B: persistent recurrence. 64 blocks x 128 thr (4 waves).
// Block owns 16 columns of H; wave owns 16 batch rows x 16 cols x 3 gates.
// Block's W_hh slice (3 gates x 16 cols x 1024 K = 96 KB bf16) lives in LDS
// for the whole time loop. Global monotonic-counter barrier each step;
// hidden state double-buffered (f32 carry + bf16 matmul copy).
// ---------------------------------------------------------------------------

__global__ __launch_bounds__(128) void gru_recurrence(
    const float* __restrict__ gi, const float* __restrict__ gf,
    const unsigned short* __restrict__ Whh, const float* __restrict__ bias_hh,
    float* __restrict__ hF, unsigned short* __restrict__ hB,
    float* __restrict__ out, float* __restrict__ hlast,
    unsigned* __restrict__ ctr)
{
  // 3 gates x 32 kTiles x 64 uint4 (one 512-elem bf16 fragment) = 96 KB
  __shared__ uint4 sW4[3 * 32 * 64];

  const int lane = threadIdx.x & 31;
  const int wave = threadIdx.x >> 5;
  const int cb   = blockIdx.x << 4;           // column base in H
  const int row0 = wave << 4;                 // batch row base
  const int kT   = H_DIM >> 5;                // 32
  const int n    = cb + (lane & 15);
  const int bb   = row0 + ((lane >> 4) << 3);

  const float bh0 = bias_hh[n];
  const float bh1 = bias_hh[H_DIM + n];
  const float bh2 = bias_hh[2 * H_DIM + n];
  const unsigned nblocks = gridDim.x;

  // ---- stage this block's W_hh fragments into LDS (once) ----
  {
    const long fb[3] = { (long)((0 * H_DIM + cb) >> 4) * kT,
                         (long)((1 * H_DIM + cb) >> 4) * kT,
                         (long)((2 * H_DIM + cb) >> 4) * kT };
#pragma unroll
    for (int g = 0; g < 3; ++g) {
      const uint4* gsrc = reinterpret_cast<const uint4*>(Whh + (size_t)fb[g] * 512);
      for (int i = threadIdx.x; i < 32 * 64; i += 128) {
        sW4[g * (32 * 64) + i] = gsrc[i];
      }
    }
  }
  __syncthreads();

  const int lq = lane * 2;                    // lane's uint4 offset within a frag

  for (int t = 0; t < S_DIM; ++t) {
    const unsigned short* hb_in  = hB + (size_t)(t & 1) * B_DIM * H_DIM;
    unsigned short*       hb_out = hB + (size_t)((t + 1) & 1) * B_DIM * H_DIM;
    const float*          hf_in  = hF + (size_t)(t & 1) * B_DIM * H_DIM;
    float*                hf_out = hF + (size_t)((t + 1) & 1) * B_DIM * H_DIM;

    // Prefetch the gate pre-activations this wave needs in the epilogue so
    // they are WGP$-resident by the time the WMMA chain retires.
#pragma unroll
    for (int v = 0; v < 8; ++v) {
      const int    bi = bb + v;
      const size_t r3 = ((size_t)bi * S_DIM + t) * (3 * H_DIM);
      const size_t r2 = ((size_t)bi * S_DIM + t) * (2 * H_DIM);
      __builtin_prefetch(&gi[r3 + n], 0, 0);
      __builtin_prefetch(&gi[r3 + H_DIM + n], 0, 0);
      __builtin_prefetch(&gi[r3 + 2 * H_DIM + n], 0, 0);
      __builtin_prefetch(&gf[r2 + n], 0, 0);
      __builtin_prefetch(&gf[r2 + H_DIM + n], 0, 0);
      __builtin_prefetch(&hf_in[(size_t)bi * H_DIM + n], 0, 0);
    }

    v8f a0 = (v8f){0,0,0,0,0,0,0,0};
    v8f a1 = (v8f){0,0,0,0,0,0,0,0};
    v8f a2 = (v8f){0,0,0,0,0,0,0,0};

#pragma unroll 4
    for (int kt = 0; kt < kT; ++kt) {
      v16bf a = load_a_frag(hb_in, H_DIM, row0, kt << 5, lane);
      FragU u0, u1, u2;
      u0.q[0] = sW4[(0 * 32 + kt) * 64 + lq];
      u0.q[1] = sW4[(0 * 32 + kt) * 64 + lq + 1];
      u1.q[0] = sW4[(1 * 32 + kt) * 64 + lq];
      u1.q[1] = sW4[(1 * 32 + kt) * 64 + lq + 1];
      u2.q[0] = sW4[(2 * 32 + kt) * 64 + lq];
      u2.q[1] = sW4[(2 * 32 + kt) * 64 + lq + 1];
      a0 = wmma_bf16(a, u0.v, a0);
      a1 = wmma_bf16(a, u1.v, a1);
      a2 = wmma_bf16(a, u2.v, a2);
    }

#pragma unroll
    for (int v = 0; v < 8; ++v) {
      const int    bi = bb + v;
      const size_t r3 = ((size_t)bi * S_DIM + t) * (3 * H_DIM);
      const size_t r2 = ((size_t)bi * S_DIM + t) * (2 * H_DIM);
      const float i_r = gi[r3 + n];
      const float i_i = gi[r3 + H_DIM + n];
      const float i_n = gi[r3 + 2 * H_DIM + n];
      const float f_r = gf[r2 + n];
      const float f_i = gf[r2 + H_DIM + n];
      const float h_r = a0[v] + bh0;
      const float h_i = a1[v] + bh1;
      const float h_n = a2[v] + bh2;
      const float rg = 1.0f / (1.0f + __expf(-(i_r + h_r + f_r)));
      const float ig = 1.0f / (1.0f + __expf(-(i_i + h_i + f_i)));
      const float ng = tanhf(i_n + rg * h_n);
      const float hp = hf_in[(size_t)bi * H_DIM + n];
      const float hy = ng + ig * (hp - ng);
      out[((size_t)bi * S_DIM + t) * H_DIM + n] = hy;
      hf_out[(size_t)bi * H_DIM + n] = hy;
      hb_out[(size_t)bi * H_DIM + n] = f2bf(hy);
      if (t == S_DIM - 1) hlast[(size_t)bi * H_DIM + n] = hy;
    }

    // ---- device-wide step barrier (monotonic counter, no reset races) ----
    __threadfence();
    __syncthreads();
    if (threadIdx.x == 0) {
      atomicAdd(ctr, 1u);
      const unsigned target = nblocks * (unsigned)(t + 1);
      while (atomicAdd(ctr, 0u) < target) { __builtin_amdgcn_s_sleep(1); }
    }
    __syncthreads();
    __threadfence();
  }
}

// ---------------------------------------------------------------------------
// Host launcher
// ---------------------------------------------------------------------------

extern "C" void kernel_launch(void* const* d_in, const int* in_sizes, int n_in,
                              void* d_out, int out_size, void* d_ws, size_t ws_size,
                              hipStream_t stream) {
  (void)in_sizes; (void)n_in; (void)out_size; (void)ws_size;
  const float* input_feats = (const float*)d_in[0];
  const float* feat0 = (const float*)d_in[1];
  const float* feat1 = (const float*)d_in[2];
  const float* w_ih  = (const float*)d_in[3];
  const float* w_hh  = (const float*)d_in[4];
  const float* w_f0  = (const float*)d_in[5];
  const float* w_f1  = (const float*)d_in[6];
  const float* b_ih  = (const float*)d_in[7];
  const float* b_hh  = (const float*)d_in[8];
  const float* b_f0  = (const float*)d_in[9];
  const float* b_f1  = (const float*)d_in[10];

  char* base = (char*)d_ws;
  size_t off = 0;
  auto alloc = [&](size_t bytes) -> void* {
    void* r = base + off;
    off += (bytes + 255) & ~(size_t)255;
    return r;
  };
  float*          gi   = (float*)alloc((size_t)BS * 3 * H_DIM * 4);      // 192 MB
  float*          gf   = (float*)alloc((size_t)BS * 2 * H_DIM * 4);      // 128 MB
  unsigned short* Xb   = (unsigned short*)alloc((size_t)BS * H_DIM * 2); //  32 MB
  unsigned short* F0b  = (unsigned short*)alloc((size_t)BS * H_DIM * 2);
  unsigned short* F1b  = (unsigned short*)alloc((size_t)BS * H_DIM * 2);
  unsigned short* WpIH = (unsigned short*)alloc((size_t)3072 * 1024 * 2);
  unsigned short* WpHH = (unsigned short*)alloc((size_t)3072 * 1024 * 2);
  unsigned short* WpF0 = (unsigned short*)alloc((size_t)2048 * 1024 * 2);
  unsigned short* WpF1 = (unsigned short*)alloc((size_t)2048 * 1024 * 2);
  float*          hF   = (float*)alloc((size_t)2 * B_DIM * H_DIM * 4);
  unsigned short* hB   = (unsigned short*)alloc((size_t)2 * B_DIM * H_DIM * 2);
  unsigned*       ctr  = (unsigned*)alloc(256);

  float* out   = (float*)d_out;                         // [B,S,H]
  float* hlast = out + (size_t)B_DIM * S_DIM * H_DIM;   // [B,H]

  init_state<<<512, 256, 0, stream>>>(hF, hB, ctr);

  const long NE = (long)BS * H_DIM;                     // 16.7M per tensor
  f32_to_bf16_k<<<4096, 256, 0, stream>>>(input_feats, Xb, NE);
  f32_to_bf16_k<<<4096, 256, 0, stream>>>(feat0, F0b, NE);
  f32_to_bf16_k<<<4096, 256, 0, stream>>>(feat1, F1b, NE);

  pack_weights<<<768, 256, 0, stream>>>(w_ih, WpIH, 3072, 1024);
  pack_weights<<<768, 256, 0, stream>>>(w_hh, WpHH, 3072, 1024);
  pack_weights<<<512, 256, 0, stream>>>(w_f0, WpF0, 2048, 1024);
  pack_weights<<<512, 256, 0, stream>>>(w_f1, WpF1, 2048, 1024);

  gemm_bf16_nt<<<dim3(24, 128), 256, 0, stream>>>(Xb, WpIH, nullptr, nullptr,
                                                  b_ih, nullptr, gi, BS, 3072, 1024);
  gemm_bf16_nt<<<dim3(16, 128), 256, 0, stream>>>(F0b, WpF0, F1b, WpF1,
                                                  b_f0, b_f1, gf, BS, 2048, 1024);

  gru_recurrence<<<64, 128, 0, stream>>>(gi, gf, WpHH, b_hh, hF, hB, out, hlast, ctr);
}
